// SWLMWA_67345087201539
// MI455X (gfx1250) — compile-verified, matrix operand order checked by reading, not run on hardware
//
#include <hip/hip_runtime.h>
#include <hip/hip_bf16.h>

typedef __bf16 bf16;
typedef __bf16 v16bf  __attribute__((ext_vector_type(16)));
typedef __bf16 bf16x8 __attribute__((ext_vector_type(8)));
typedef float  v8f    __attribute__((ext_vector_type(8)));

#define DIMC    256
#define NHEADS  8
#define HD      32
#define HH      70
#define WIMG    220
#define NTOK    (HH*WIMG)        // 15400
#define BATCH   8
#define ROWS    (BATCH*NTOK)     // 123200
#define WH      7
#define WWIN    11
#define WS      77
#define NWW     20
#define NWIN_B  200
#define NWIN    (BATCH*NWIN_B)   // 1600
#define SH      3
#define SW      5
#define HIDDEN  512
#define EPSV    1e-5f

// ---- CDNA5 async global->LDS path (guarded; sync fallback keeps compile safe)
#if defined(__has_builtin)
#  if __has_builtin(__builtin_amdgcn_global_load_async_to_lds_b128) && \
      __has_builtin(__builtin_amdgcn_s_wait_asynccnt)
#    define USE_ASYNC 1
#  endif
#endif
#ifndef USE_ASYNC
#  define USE_ASYNC 0
#endif

#if USE_ASYNC
typedef int v4i __attribute__((vector_size(16)));
typedef __attribute__((address_space(1))) v4i* gv4p;   // global int4*
typedef __attribute__((address_space(3))) v4i* lv4p;   // LDS int4*
#endif

static __device__ __forceinline__ bf16 f2bf(float x) { return (bf16)x; }

static __device__ __forceinline__ void copy16_async(const bf16* g, bf16* l) {
#if USE_ASYNC
    __builtin_amdgcn_global_load_async_to_lds_b128(
        (gv4p)const_cast<bf16*>(g), (lv4p)l, 0, 0);
#else
    *(bf16x8*)l = *(const bf16x8*)g;
#endif
}
static __device__ __forceinline__ void wait_async_4() {
#if USE_ASYNC
    __builtin_amdgcn_s_wait_asynccnt(4);
#endif
}
static __device__ __forceinline__ void wait_async_0() {
#if USE_ASYNC
    __builtin_amdgcn_s_wait_asynccnt(0);
#endif
}

static __device__ __forceinline__ v8f zero8() {
    v8f z;
#pragma unroll
    for (int i = 0; i < 8; ++i) z[i] = 0.f;
    return z;
}

// Load one 16x32 bf16 WMMA fragment (A or B role) from LDS, row-major rows.
// Lane half h reads K = 8h..8h+7 (VGPR0-3) and K = 16+8h.. (VGPR4-7).
static __device__ __forceinline__ v16bf load_frag(const bf16* rowp, int h) {
    bf16x8 lo = *(const bf16x8*)(rowp + 8 * h);
    bf16x8 hi = *(const bf16x8*)(rowp + 16 + 8 * h);
    v16bf a;
#pragma unroll
    for (int i = 0; i < 8; ++i) { a[i] = lo[i]; a[i + 8] = hi[i]; }
    return a;
}

static __device__ __forceinline__ float blockSum(float v, float* red) {
#pragma unroll
    for (int o = 16; o; o >>= 1) v += __shfl_xor(v, o, 32);
    int w = threadIdx.x >> 5, l = threadIdx.x & 31;
    if (l == 0) red[w] = v;
    __syncthreads();
    if (w == 0) {
        float t = (l < 8) ? red[l] : 0.f;
#pragma unroll
        for (int o = 4; o; o >>= 1) t += __shfl_xor(t, o, 32);
        if (l == 0) red[0] = t;
    }
    __syncthreads();
    float r = red[0];
    __syncthreads();
    return r;
}

// --- weight convert: f32 [K][N] row-major  ->  bf16 [N][K] (transposed) ---
__global__ __launch_bounds__(256) void convert_wt(const float* __restrict__ w,
                                                  bf16* __restrict__ wt,
                                                  int K, int N) {
    int idx = blockIdx.x * 256 + threadIdx.x;
    if (idx >= K * N) return;
    int n = idx / K, k = idx - n * K;
    wt[idx] = f2bf(w[(size_t)k * N + n]);
}

// --- LayerNorm; PERM=true gathers through the roll+window permutation ---
template <bool PERM>
__global__ __launch_bounds__(256) void ln_kernel(const float* __restrict__ x,
                                                 const float* __restrict__ g,
                                                 const float* __restrict__ bb,
                                                 bf16* __restrict__ out) {
    __shared__ float red[8];
    int r = blockIdx.x;
    size_t src;
    if (PERM) {
        int win = r / WS, t = r - win * WS;
        int b = win / NWIN_B, w2 = win - b * NWIN_B;
        int wh = w2 / NWW, wn = w2 - wh * NWW;
        int i = wh * WH + t / WWIN;
        int j = wn * WWIN + (t - (t / WWIN) * WWIN);
        int si = i + SH; if (si >= HH) si -= HH;
        int sj = j + SW; if (sj >= WIMG) sj -= WIMG;
        src = ((size_t)b * NTOK + si * WIMG + sj) * DIMC;
    } else {
        src = (size_t)r * DIMC;
    }
    int c = threadIdx.x;
    float v = x[src + c];
    float mu = blockSum(v, red) * (1.f / DIMC);
    float d = v - mu;
    float var = blockSum(d * d, red) * (1.f / DIMC);
    float rs = rsqrtf(var + EPSV);
    out[(size_t)r * DIMC + c] = f2bf(d * rs * g[c] + bb[c]);
}

// --- bf16 WMMA GEMM, 128x128 block tile, double-buffered async LDS staging.
// C = A[M,K] @ Wt[N,K]^T + bias, epilogue by MODE:
// MODE 0: bf16 store (qkv)           MODE 1: f32 scatter + residual (proj)
// MODE 2: GELU -> bf16 (fc1)         MODE 3: f32 + residual -> out (fc2)
// NOTE: out-of-range M rows are loaded unguarded (reads stay inside d_ws) and
// their results are discarded by the guarded epilogue stores.
template <int MODE>
__global__ __launch_bounds__(256) void gemm_kernel(const bf16* __restrict__ A,
                                                   const bf16* __restrict__ Wt,
                                                   const float* __restrict__ bias,
                                                   void* __restrict__ outp,
                                                   const float* __restrict__ resid,
                                                   int Mrows, int K, int Nout) {
    __shared__ bf16 As[2][128 * 40];  // stride 40 halves: bank-conflict pad
    __shared__ bf16 Bs[2][128 * 40];
    int tid = threadIdx.x;
    int Mbase = blockIdx.x * 128;
    int Nbase = blockIdx.y * 128;
    int wave = tid >> 5, lane = tid & 31;
    int wm = wave >> 1, wn = wave & 1;          // 4 x 2 waves -> 32x64 per wave
    int lm = lane & 15, h = lane >> 4;

    int r0 = tid >> 2, g0 = (tid & 3) * 8;      // staging coords (2 ops each)
    int r1 = (tid + 256) >> 2, g1 = g0;

    v8f acc[2][4];
#pragma unroll
    for (int i = 0; i < 2; ++i)
#pragma unroll
        for (int j = 0; j < 4; ++j) acc[i][j] = zero8();

    int nch = K >> 5;
    // prologue: stage chunk 0 into buffer 0
    copy16_async(A  + (size_t)(Mbase + r0) * K + g0, &As[0][r0 * 40 + g0]);
    copy16_async(A  + (size_t)(Mbase + r1) * K + g1, &As[0][r1 * 40 + g1]);
    copy16_async(Wt + (size_t)(Nbase + r0) * K + g0, &Bs[0][r0 * 40 + g0]);
    copy16_async(Wt + (size_t)(Nbase + r1) * K + g1, &Bs[0][r1 * 40 + g1]);

    for (int kc = 0; kc < nch; ++kc) {
        int cur = kc & 1;
        if (kc + 1 < nch) {                     // stage next chunk into other buf
            int ko = (kc + 1) * 32;
            copy16_async(A  + (size_t)(Mbase + r0) * K + ko + g0, &As[cur ^ 1][r0 * 40 + g0]);
            copy16_async(A  + (size_t)(Mbase + r1) * K + ko + g1, &As[cur ^ 1][r1 * 40 + g1]);
            copy16_async(Wt + (size_t)(Nbase + r0) * K + ko + g0, &Bs[cur ^ 1][r0 * 40 + g0]);
            copy16_async(Wt + (size_t)(Nbase + r1) * K + ko + g1, &Bs[cur ^ 1][r1 * 40 + g1]);
            wait_async_4();                     // current chunk's 4 ops complete
        } else {
            wait_async_0();
        }
        __syncthreads();                        // all waves' staging visible
        v16bf a0 = load_frag(&As[cur][(wm * 32 + lm) * 40], h);
        v16bf a1 = load_frag(&As[cur][(wm * 32 + 16 + lm) * 40], h);
#pragma unroll
        for (int j = 0; j < 4; ++j) {
            v16bf b = load_frag(&Bs[cur][(wn * 64 + j * 16 + lm) * 40], h);
            acc[0][j] = __builtin_amdgcn_wmma_f32_16x16x32_bf16(false, a0, false, b, (short)0, acc[0][j], false, false);
            acc[1][j] = __builtin_amdgcn_wmma_f32_16x16x32_bf16(false, a1, false, b, (short)0, acc[1][j], false, false);
        }
        __syncthreads();                        // done reading before overwrite
    }

    // epilogue: C layout lane -> (n = lane%16), row m = 8*h + r
#pragma unroll
    for (int ti = 0; ti < 2; ++ti) {
#pragma unroll
        for (int tj = 0; tj < 4; ++tj) {
            const v8f& c = acc[ti][tj];
            int col = Nbase + wn * 64 + tj * 16 + lm;
            float bc = bias[col];
#pragma unroll
            for (int r = 0; r < 8; ++r) {
                int row = Mbase + wm * 32 + ti * 16 + h * 8 + r;
                if (row >= Mrows) continue;
                float v = c[r] + bc;
                if (MODE == 0) {
                    ((bf16*)outp)[(size_t)row * Nout + col] = f2bf(v);
                } else if (MODE == 2) {
                    float gv = 0.5f * v * (1.f + erff(v * 0.70710678118654752f));
                    ((bf16*)outp)[(size_t)row * Nout + col] = f2bf(gv);
                } else if (MODE == 3) {
                    float* o = (float*)outp;
                    size_t d = (size_t)row * Nout + col;
                    o[d] = resid[d] + v;
                } else {  // MODE 1: scatter window-row -> token position (+roll)
                    int win = row / WS, t = row - win * WS;
                    int b = win / NWIN_B, w2 = win - b * NWIN_B;
                    int wh = w2 / NWW, wnc = w2 - wh * NWW;
                    int ih = t / WWIN, iw = t - ih * WWIN;
                    int i = wh * WH + ih, j = wnc * WWIN + iw;
                    int si = i + SH; if (si >= HH) si -= HH;
                    int sj = j + SW; if (sj >= WIMG) sj -= WIMG;
                    size_t d = ((size_t)b * NTOK + si * WIMG + sj) * DIMC + col;
                    ((float*)outp)[d] = resid[d] + v;
                }
            }
        }
    }
}

// --- Windowed attention: one block (4 waves) per (window, head) ---
__global__ __launch_bounds__(128) void attn_kernel(const bf16* __restrict__ qkv,
                                                   bf16* __restrict__ out) {
    __shared__ bf16 Qs[80 * 32];
    __shared__ bf16 Ks[80 * 32];
    __shared__ bf16 Vt[32 * 96];   // V transposed: [n=32][k padded to 96]
    __shared__ bf16 Ps[80 * 96];   // softmax probs, K padded to 96
    __shared__ float Ss[80 * 80];  // scores
    int blk = blockIdx.x;
    int win = blk >> 3, head = blk & 7;
    int tid = threadIdx.x;

    for (int idx = tid; idx < 80 * 4; idx += 128) {   // stage Q,K,V (zero-pad)
        int row = idx >> 2, g = idx & 3;
        bf16x8 q, k, v;
        if (row < WS) {
            size_t base = ((size_t)(win * WS + row)) * 768 + head * 32 + g * 8;
            q = *(const bf16x8*)(qkv + base);
            k = *(const bf16x8*)(qkv + base + 256);
            v = *(const bf16x8*)(qkv + base + 512);
        } else {
#pragma unroll
            for (int e = 0; e < 8; ++e) { q[e] = (bf16)0.f; k[e] = (bf16)0.f; v[e] = (bf16)0.f; }
        }
        *(bf16x8*)(Qs + row * 32 + g * 8) = q;
        *(bf16x8*)(Ks + row * 32 + g * 8) = k;
#pragma unroll
        for (int e = 0; e < 8; ++e) Vt[(g * 8 + e) * 96 + row] = v[e];
    }
    for (int idx = tid; idx < 32 * 16; idx += 128)    // zero Vt K-tail 80..95
        Vt[(idx >> 4) * 96 + 80 + (idx & 15)] = (bf16)0.f;
    for (int idx = tid; idx < 3 * 96; idx += 128)     // zero P pad rows 77..79
        Ps[(WS + idx / 96) * 96 + idx % 96] = (bf16)0.f;
    __syncthreads();

    int wave = tid >> 5, lane = tid & 31, lm = lane & 15, h = lane >> 4;
    const float scale = 0.17677669529663687f;  // 1/sqrt(32)
    for (int t = wave; t < 25; t += 4) {       // S = Q @ K^T : 5x5 tiles, K=32
        int mt = t / 5, nt = t - mt * 5;
        v16bf a = load_frag(Qs + (mt * 16 + lm) * 32, h);
        v16bf b = load_frag(Ks + (nt * 16 + lm) * 32, h);
        v8f c = zero8();
        c = __builtin_amdgcn_wmma_f32_16x16x32_bf16(false, a, false, b, (short)0, c, false, false);
        int col = nt * 16 + lm;
#pragma unroll
        for (int r = 0; r < 8; ++r) {
            float s = c[r] * scale;
            if (col >= WS) s = -1e9f;
            Ss[(mt * 16 + h * 8 + r) * 80 + col] = s;
        }
    }
    __syncthreads();

    if (tid < WS) {                             // softmax over 77 valid cols
        float* srow = Ss + tid * 80;
        float mx = srow[0];
        for (int j = 1; j < WS; ++j) mx = fmaxf(mx, srow[j]);
        float sum = 0.f;
        for (int j = 0; j < WS; ++j) { float e = __expf(srow[j] - mx); srow[j] = e; sum += e; }
        float inv = 1.f / sum;
        bf16* prow = Ps + tid * 96;
        for (int j = 0; j < WS; ++j) prow[j] = f2bf(srow[j] * inv);
        for (int j = WS; j < 96; ++j) prow[j] = (bf16)0.f;
    }
    __syncthreads();

    for (int t = wave; t < 10; t += 4) {        // O = P @ V : 5x2 tiles, K=96
        int mt = t >> 1, nt = t & 1;
        v8f c = zero8();
#pragma unroll
        for (int kc = 0; kc < 3; ++kc) {
            v16bf a = load_frag(Ps + (mt * 16 + lm) * 96 + kc * 32, h);
            v16bf b = load_frag(Vt + (nt * 16 + lm) * 96 + kc * 32, h);
            c = __builtin_amdgcn_wmma_f32_16x16x32_bf16(false, a, false, b, (short)0, c, false, false);
        }
#pragma unroll
        for (int r = 0; r < 8; ++r) {
            int row = mt * 16 + h * 8 + r;
            if (row < WS) {
                size_t d = ((size_t)(win * WS + row)) * DIMC + head * 32 + nt * 16 + lm;
                out[d] = f2bf(c[r]);
            }
        }
    }
}

extern "C" void kernel_launch(void* const* d_in, const int* in_sizes, int n_in,
                              void* d_out, int out_size, void* d_ws, size_t ws_size,
                              hipStream_t stream) {
    (void)in_sizes; (void)n_in; (void)out_size; (void)ws_size;
    const float* x     = (const float*)d_in[0];
    const float* ln1g  = (const float*)d_in[3];
    const float* ln1b  = (const float*)d_in[4];
    const float* qkvw  = (const float*)d_in[5];
    const float* qkvb  = (const float*)d_in[6];
    const float* projw = (const float*)d_in[7];
    const float* projb = (const float*)d_in[8];
    const float* ln2g  = (const float*)d_in[9];
    const float* ln2b  = (const float*)d_in[10];
    const float* fc1w  = (const float*)d_in[11];
    const float* fc1b  = (const float*)d_in[12];
    const float* fc2w  = (const float*)d_in[13];
    const float* fc2b  = (const float*)d_in[14];

    char* ws = (char*)d_ws;
    size_t off = 0;
    auto alloc = [&](size_t bytes) -> void* {
        void* p = ws + off;
        off += (bytes + 255) & ~(size_t)255;
        return p;
    };
    bf16*  xnw    = (bf16*)alloc((size_t)ROWS * DIMC * 2);
    bf16*  qkv    = (bf16*)alloc((size_t)ROWS * 768 * 2);
    bf16*  attn   = (bf16*)alloc((size_t)ROWS * DIMC * 2);
    float* x1     = (float*)alloc((size_t)ROWS * DIMC * 4);
    bf16*  qkvwt  = (bf16*)alloc((size_t)768 * 256 * 2);
    bf16*  projwt = (bf16*)alloc((size_t)256 * 256 * 2);
    bf16*  fc1wt  = (bf16*)alloc((size_t)512 * 256 * 2);
    bf16*  fc2wt  = (bf16*)alloc((size_t)256 * 512 * 2);
    bf16*  xn2    = xnw;  // dead after qkv GEMM -> reuse
    bf16*  hidden = qkv;  // dead after attention -> reuse

    convert_wt<<<dim3((768 * 256 + 255) / 256), 256, 0, stream>>>(qkvw, qkvwt, 256, 768);
    convert_wt<<<dim3((256 * 256 + 255) / 256), 256, 0, stream>>>(projw, projwt, 256, 256);
    convert_wt<<<dim3((512 * 256 + 255) / 256), 256, 0, stream>>>(fc1w, fc1wt, 256, 512);
    convert_wt<<<dim3((256 * 512 + 255) / 256), 256, 0, stream>>>(fc2w, fc2wt, 512, 256);

    ln_kernel<true><<<dim3(ROWS), 256, 0, stream>>>(x, ln1g, ln1b, xnw);
    gemm_kernel<0><<<dim3(963, 6), 256, 0, stream>>>(xnw, qkvwt, qkvb, (void*)qkv, nullptr, ROWS, 256, 768);
    attn_kernel<<<dim3(NWIN * NHEADS), 128, 0, stream>>>(qkv, attn);
    gemm_kernel<1><<<dim3(963, 2), 256, 0, stream>>>(attn, projwt, projb, (void*)x1, x, ROWS, 256, 256);
    ln_kernel<false><<<dim3(ROWS), 256, 0, stream>>>(x1, ln2g, ln2b, xn2);
    gemm_kernel<2><<<dim3(963, 4), 256, 0, stream>>>(xn2, fc1wt, fc1b, (void*)hidden, nullptr, ROWS, 256, 512);
    gemm_kernel<3><<<dim3(963, 2), 256, 0, stream>>>(hidden, fc2wt, fc2b, d_out, x1, ROWS, 512, 256);
}
